// GAT_20091857011053
// MI455X (gfx1250) — matrix operand order repaired and checked by reading, not compiled, not run on hardware
//
#include <hip/hip_runtime.h>
#include <hip/hip_bf16.h>

#define NN   50000
#define EE   800000
#define ET   850000   // EE + NN self loops
#define FIN  767
#define HID  256
#define KT1  24       // 768 / 32 K-steps for layer 1
#define KT2  8        // 256 / 32 K-steps for layer 2

typedef __attribute__((ext_vector_type(16))) __bf16        v16bf;
typedef __attribute__((ext_vector_type(8)))  float         v8f;
typedef __attribute__((ext_vector_type(4)))  float         f32x4;
typedef __attribute__((ext_vector_type(4)))  unsigned int  u32x4;
typedef __attribute__((ext_vector_type(2)))  unsigned int  u32x2;

union FragU {
    v16bf          v;
    unsigned short u[16];
    u32x4          q[2];
};

// ---------- helpers ----------
__device__ __forceinline__ unsigned short f2bf(float f) {
    unsigned int u = __float_as_uint(f);
    u += 0x7fffu + ((u >> 16) & 1u);           // round-to-nearest-even
    return (unsigned short)(u >> 16);
}
__device__ __forceinline__ unsigned fkey(float f) {     // monotone float -> uint
    unsigned u = __float_as_uint(f);
    return (u & 0x80000000u) ? ~u : (u | 0x80000000u);
}
__device__ __forceinline__ float kdecode(unsigned k) {
    unsigned u = (k & 0x80000000u) ? (k & 0x7fffffffu) : ~k;
    return __uint_as_float(u);
}
__device__ __forceinline__ void edge_sd(const int* __restrict__ ei, int e, int& s, int& d) {
    if (e < EE) { s = ei[e]; d = ei[EE + e]; } else { s = e - EE; d = e - EE; }
}

// ---------- weight pre-swizzle into WMMA B-fragment-linear bf16 ----------
// layout: [ktile][ctile][lane][16 bf16], lane's 16 values are K = kbase..kbase+7, kbase+16..kbase+23
__global__ __launch_bounds__(256) void gat_swizzW1(const float* __restrict__ W1,
                                                   unsigned short* __restrict__ W1b) {
    int id = blockIdx.x * 256 + threadIdx.x;
    if (id >= KT1 * 16 * 32 * 16) return;
    int i    = id & 15;
    int lane = (id >> 4) & 31;
    int ct   = (id >> 9) & 15;
    int kt   = id >> 13;
    int col  = ct * 16 + (lane & 15);
    int k    = kt * 32 + ((lane < 16) ? 0 : 8) + ((i < 8) ? i : i + 8);
    float v  = (k < FIN) ? W1[k * HID + col] : 0.f;
    W1b[id]  = f2bf(v);
}

__global__ __launch_bounds__(256) void gat_swizzW2(const float* __restrict__ W2,
                                                   unsigned short* __restrict__ W2b) {
    int id = blockIdx.x * 256 + threadIdx.x;
    if (id >= KT2 * 32 * 16) return;
    int i    = id & 15;
    int lane = (id >> 4) & 31;
    int kt   = id >> 9;
    int col  = lane & 15;
    int k    = kt * 32 + ((lane < 16) ? 0 : 8) + ((i < 8) ? i : i + 8);
    float v  = (col < 10) ? W2[k * 10 + col] : 0.f;
    W2b[id]  = f2bf(v);
}

// ---------- per-call initialization (bias folded into accumulators) ----------
__global__ __launch_bounds__(256) void gat_initOut1(float* __restrict__ o, const float* __restrict__ b1) {
    int i = blockIdx.x * 256 + threadIdx.x;
    if (i < NN * HID) o[i] = b1[i & 255];
}
__global__ __launch_bounds__(256) void gat_initSc1(unsigned* __restrict__ mk, float* __restrict__ dn) {
    int i = blockIdx.x * 256 + threadIdx.x;
    if (i < NN * 8) { mk[i] = 0u; dn[i] = 0.f; }
}
__global__ __launch_bounds__(256) void gat_initOut2(float* __restrict__ o, const float* __restrict__ b2) {
    int i = blockIdx.x * 256 + threadIdx.x;
    if (i < NN * 10) o[i] = b2[i % 10];
}
__global__ __launch_bounds__(256) void gat_initSc2(unsigned* __restrict__ mk, float* __restrict__ dn) {
    int i = blockIdx.x * 256 + threadIdx.x;
    if (i < NN) { mk[i] = 0u; dn[i] = 0.f; }
}

// ---------- layer 1 GEMM: h1[N,256] = bf16(x) @ bf16(W1), WMMA ----------
// block = 8 waves; block covers 32 rows x 256 cols; wave = 16 rows x 64 cols.
// A tile: fp32->bf16 converted cooperatively into LDS (ds_store_b64).
// B panel: 16 KB/K-step staged via async global->LDS DMA (ASYNCcnt), bypassing VGPRs.
// Double-buffered, one barrier per K-step; DMA + staging overlap the WMMAs.
__global__ __launch_bounds__(256) void gat_gemm1(const float* __restrict__ x,
                                                 const unsigned short* __restrict__ W1b,
                                                 float* __restrict__ h1) {
    __shared__ __align__(16) unsigned short Abuf[2][32][32];    // 2 x 2 KB
    __shared__ __align__(16) unsigned short Bbuf[2][16 * 512];  // 2 x 16 KB
    const int tid    = threadIdx.x;
    const int lane   = tid & 31;
    const int wv     = tid >> 5;
    const int r0     = blockIdx.x * 32;
    const int rt     = (wv >> 2) * 16;      // 0 or 16
    const int colGrp = wv & 3;              // 4 col-tiles each
    const int arow   = rt + (lane & 15);
    const int ks     = (lane < 16) ? 0 : 8;

    // staging: each thread converts 4 consecutive floats -> one ds_store_b64
    const int  rr   = tid >> 3;             // row within tile 0..31
    const int  kk0  = (tid & 7) * 4;        // K offset within tile
    const bool tail = (r0 + 32 > NN);       // block-uniform
    const int  gr   = r0 + rr;

    auto stageA = [&](int kt, int buf) {
        int kb = kt * 32 + kk0;
        float f0, f1, f2, f3;
        if (!tail && kt < KT1 - 1) {        // wave-uniform fast path
            const float* p = x + gr * FIN + kb;
            f0 = p[0]; f1 = p[1]; f2 = p[2]; f3 = p[3];
        } else {
            f0 = (gr < NN && kb + 0 < FIN) ? x[gr * FIN + kb + 0] : 0.f;
            f1 = (gr < NN && kb + 1 < FIN) ? x[gr * FIN + kb + 1] : 0.f;
            f2 = (gr < NN && kb + 2 < FIN) ? x[gr * FIN + kb + 2] : 0.f;
            f3 = (gr < NN && kb + 3 < FIN) ? x[gr * FIN + kb + 3] : 0.f;
        }
        unsigned lo = (unsigned)f2bf(f0) | ((unsigned)f2bf(f1) << 16);
        unsigned hi = (unsigned)f2bf(f2) | ((unsigned)f2bf(f3) << 16);
        u32x2 pk = {lo, hi};
        *(u32x2*)&Abuf[buf][rr][kk0] = pk;
    };

    // async DMA of the bf16 B K-panel (16 KB) straight into LDS (no VGPR round trip)
    auto stageB = [&](int kt, int buf) {
        const char* gsrc = (const char*)(W1b + (size_t)kt * 16 * 512) + tid * 64;
        unsigned    lds  = (unsigned)(uintptr_t)&Bbuf[buf][0] + (unsigned)tid * 64u;
        #pragma unroll
        for (int j = 0; j < 4; ++j) {
            unsigned long long ga = (unsigned long long)(uintptr_t)(gsrc + j * 16);
            unsigned           la = lds + (unsigned)(j * 16);
            asm volatile("global_load_async_to_lds_b128 %0, %1, off"
                         :: "v"(la), "v"(ga) : "memory");
        }
    };

    v8f acc[4] = {};
    stageA(0, 0);
    stageB(0, 0);
    asm volatile("s_wait_asynccnt 0x0" ::: "memory");
    __syncthreads();

    for (int kt = 0; kt < KT1; ++kt) {
        const int cur = kt & 1;

        FragU a;
        a.q[0] = *(const u32x4*)&Abuf[cur][arow][ks];
        a.q[1] = *(const u32x4*)&Abuf[cur][arow][ks + 16];

        const unsigned short* bp = &Bbuf[cur][(colGrp * 4) * 512 + lane * 16];
        FragU b[4];
        #pragma unroll
        for (int ct = 0; ct < 4; ++ct) {
            b[ct].q[0] = *(const u32x4*)(bp + ct * 512);
            b[ct].q[1] = *(const u32x4*)(bp + ct * 512 + 8);
        }

        if (kt + 1 < KT1) {                 // overlap next-tile DMA/stage with WMMAs
            stageA(kt + 1, cur ^ 1);
            stageB(kt + 1, cur ^ 1);
        }

        #pragma unroll
        for (int ct = 0; ct < 4; ++ct)
            acc[ct] = __builtin_amdgcn_wmma_f32_16x16x32_bf16(
                false, a.v, false, b[ct].v, (short)0, acc[ct], false, false);

        asm volatile("s_wait_asynccnt 0x0" ::: "memory");
        __syncthreads();
    }

    const int rhalf = lane >> 4;
    #pragma unroll
    for (int ct = 0; ct < 4; ++ct) {
        int colb = (colGrp * 4 + ct) * 16 + (lane & 15);
        #pragma unroll
        for (int v = 0; v < 8; ++v) {
            int grs = r0 + rt + v + 8 * rhalf;
            if (grs < NN) h1[grs * HID + colb] = acc[ct][v];
        }
    }
}

// ---------- layer 1 attention coefficients ----------
__global__ __launch_bounds__(256) void gat_attn1(const float* __restrict__ h1,
                                                 const float* __restrict__ att_s,
                                                 const float* __restrict__ att_d,
                                                 float* __restrict__ a_s,
                                                 float* __restrict__ a_d) {
    int id = blockIdx.x * 256 + threadIdx.x;
    if (id >= NN * 8) return;
    int n = id >> 3, hh = id & 7;
    const f32x4* hp  = (const f32x4*)(h1 + (size_t)n * HID + hh * 32);
    const f32x4* sp  = (const f32x4*)(att_s + hh * 32);
    const f32x4* dp  = (const f32x4*)(att_d + hh * 32);
    float s = 0.f, d = 0.f;
    #pragma unroll
    for (int c4 = 0; c4 < 8; ++c4) {
        f32x4 hv = hp[c4], sv = sp[c4], dv = dp[c4];
        #pragma unroll
        for (int j = 0; j < 4; ++j) { s += hv[j] * sv[j]; d += hv[j] * dv[j]; }
    }
    a_s[id] = s;
    a_d[id] = d;
}

__global__ __launch_bounds__(256) void gat_edgemax1(const int* __restrict__ ei,
                                                    const float* __restrict__ a_s,
                                                    const float* __restrict__ a_d,
                                                    unsigned* __restrict__ mk) {
    int e = blockIdx.x * 256 + threadIdx.x;
    if (e >= ET) return;
    int s, d; edge_sd(ei, e, s, d);
    #pragma unroll
    for (int hh = 0; hh < 8; ++hh) {
        float v = a_s[s * 8 + hh] + a_d[d * 8 + hh];
        v = v > 0.f ? v : 0.2f * v;
        atomicMax(&mk[d * 8 + hh], fkey(v));
    }
}

__global__ __launch_bounds__(256) void gat_edgesum1(const int* __restrict__ ei,
                                                    const float* __restrict__ a_s,
                                                    const float* __restrict__ a_d,
                                                    const unsigned* __restrict__ mk,
                                                    float* __restrict__ dn) {
    int e = blockIdx.x * 256 + threadIdx.x;
    if (e >= ET) return;
    int s, d; edge_sd(ei, e, s, d);
    #pragma unroll
    for (int hh = 0; hh < 8; ++hh) {
        float v = a_s[s * 8 + hh] + a_d[d * 8 + hh];
        v = v > 0.f ? v : 0.2f * v;
        atomicAdd(&dn[d * 8 + hh], __expf(v - kdecode(mk[d * 8 + hh])));
    }
}

// one wave per edge; lanes 0..7 compute per-head alpha, shfl-broadcast to all
__global__ __launch_bounds__(256) void gat_agg1(const int* __restrict__ ei,
                                                const float* __restrict__ a_s,
                                                const float* __restrict__ a_d,
                                                const unsigned* __restrict__ mk,
                                                const float* __restrict__ dn,
                                                const float* __restrict__ h1,
                                                float* __restrict__ out1) {
    int lane = threadIdx.x & 31, wv = threadIdx.x >> 5;
    int e = blockIdx.x * 8 + wv;
    if (e >= ET) return;
    int s, d; edge_sd(ei, e, s, d);
    float myAlpha = 0.f;
    if (lane < 8) {
        float v = a_s[s * 8 + lane] + a_d[d * 8 + lane];
        v = v > 0.f ? v : 0.2f * v;
        myAlpha = __expf(v - kdecode(mk[d * 8 + lane])) / (dn[d * 8 + lane] + 1e-16f);
    }
    #pragma unroll
    for (int j = 0; j < 8; ++j) {
        float alpha = __shfl(myAlpha, j, 32);
        float msg = alpha * h1[(size_t)s * HID + j * 32 + lane];
        atomicAdd(&out1[(size_t)d * HID + j * 32 + lane], msg);
    }
}

__global__ __launch_bounds__(256) void gat_elu(float* __restrict__ o) {
    int i = blockIdx.x * 256 + threadIdx.x;
    if (i >= NN * HID) return;
    float v = o[i];
    o[i] = v > 0.f ? v : (__expf(v) - 1.f);
}

// ---------- layer 2 GEMM: h2[N,10] via one padded 16-col WMMA tile ----------
__global__ __launch_bounds__(256) void gat_gemm2(const float* __restrict__ x2,
                                                 const unsigned short* __restrict__ W2b,
                                                 float* __restrict__ h2) {
    int lane = threadIdx.x & 31, wv = threadIdx.x >> 5;
    int t = blockIdx.x * 8 + wv;            // 16-row tile index, 3125 total
    if (t >= NN / 16) return;
    v8f acc = {};
    int row = t * 16 + (lane & 15);
    int ks  = (lane < 16) ? 0 : 8;
    for (int kt = 0; kt < KT2; ++kt) {
        const float* xp = x2 + (size_t)row * HID + kt * 32 + ks;
        if (kt + 1 < KT2) __builtin_prefetch(xp + 32, 0, 0);
        f32x4 v0 = *(const f32x4*)(xp);
        f32x4 v1 = *(const f32x4*)(xp + 4);
        f32x4 v2 = *(const f32x4*)(xp + 16);
        f32x4 v3 = *(const f32x4*)(xp + 20);
        FragU a;
        #pragma unroll
        for (int i = 0; i < 4; ++i) {
            a.u[i]      = f2bf(v0[i]);
            a.u[4 + i]  = f2bf(v1[i]);
            a.u[8 + i]  = f2bf(v2[i]);
            a.u[12 + i] = f2bf(v3[i]);
        }
        FragU b;
        const unsigned short* bp = W2b + ((size_t)kt * 32 + lane) * 16;
        b.q[0] = *(const u32x4*)bp;
        b.q[1] = *(const u32x4*)(bp + 8);
        acc = __builtin_amdgcn_wmma_f32_16x16x32_bf16(
            false, a.v, false, b.v, (short)0, acc, false, false);
    }
    int col = lane & 15;
    if (col < 10) {
        #pragma unroll
        for (int v = 0; v < 8; ++v) {
            int gr = t * 16 + v + 8 * (lane >> 4);
            h2[gr * 10 + col] = acc[v];
        }
    }
}

__global__ __launch_bounds__(256) void gat_attn2(const float* __restrict__ h2,
                                                 const float* __restrict__ att_s,
                                                 const float* __restrict__ att_d,
                                                 float* __restrict__ a_s,
                                                 float* __restrict__ a_d) {
    int n = blockIdx.x * 256 + threadIdx.x;
    if (n >= NN) return;
    float s = 0.f, d = 0.f;
    #pragma unroll
    for (int c = 0; c < 10; ++c) {
        float v = h2[n * 10 + c];
        s += v * att_s[c];
        d += v * att_d[c];
    }
    a_s[n] = s;
    a_d[n] = d;
}

__global__ __launch_bounds__(256) void gat_edgemax2(const int* __restrict__ ei,
                                                    const float* __restrict__ a_s,
                                                    const float* __restrict__ a_d,
                                                    unsigned* __restrict__ mk) {
    int e = blockIdx.x * 256 + threadIdx.x;
    if (e >= ET) return;
    int s, d; edge_sd(ei, e, s, d);
    float v = a_s[s] + a_d[d];
    v = v > 0.f ? v : 0.2f * v;
    atomicMax(&mk[d], fkey(v));
}

__global__ __launch_bounds__(256) void gat_edgesum2(const int* __restrict__ ei,
                                                    const float* __restrict__ a_s,
                                                    const float* __restrict__ a_d,
                                                    const unsigned* __restrict__ mk,
                                                    float* __restrict__ dn) {
    int e = blockIdx.x * 256 + threadIdx.x;
    if (e >= ET) return;
    int s, d; edge_sd(ei, e, s, d);
    float v = a_s[s] + a_d[d];
    v = v > 0.f ? v : 0.2f * v;
    atomicAdd(&dn[d], __expf(v - kdecode(mk[d])));
}

__global__ __launch_bounds__(256) void gat_agg2(const int* __restrict__ ei,
                                                const float* __restrict__ a_s,
                                                const float* __restrict__ a_d,
                                                const unsigned* __restrict__ mk,
                                                const float* __restrict__ dn,
                                                const float* __restrict__ h2,
                                                float* __restrict__ out) {
    int e = blockIdx.x * 256 + threadIdx.x;
    if (e >= ET) return;
    int s, d; edge_sd(ei, e, s, d);
    float v = a_s[s] + a_d[d];
    v = v > 0.f ? v : 0.2f * v;
    float alpha = __expf(v - kdecode(mk[d])) / (dn[d] + 1e-16f);
    #pragma unroll
    for (int c = 0; c < 10; ++c)
        atomicAdd(&out[d * 10 + c], alpha * h2[s * 10 + c]);
}

// ---------- host launch ----------
extern "C" void kernel_launch(void* const* d_in, const int* in_sizes, int n_in,
                              void* d_out, int out_size, void* d_ws, size_t ws_size,
                              hipStream_t stream) {
    const float* x   = (const float*)d_in[0];
    const int*   ei  = (const int*)d_in[1];
    const float* W1  = (const float*)d_in[2];
    const float* as1 = (const float*)d_in[3];
    const float* ad1 = (const float*)d_in[4];
    const float* b1  = (const float*)d_in[5];
    const float* W2  = (const float*)d_in[6];
    const float* as2 = (const float*)d_in[7];
    const float* ad2 = (const float*)d_in[8];
    const float* b2  = (const float*)d_in[9];
    float* out = (float*)d_out;

    char* wsb = (char*)d_ws;
    size_t o = 0;
    auto carve = [&](size_t bytes) -> void* {
        void* p = wsb + o;
        o += (bytes + 255) & ~(size_t)255;
        return p;
    };
    unsigned short* W1b = (unsigned short*)carve((size_t)KT1 * 16 * 32 * 16 * 2);
    unsigned short* W2b = (unsigned short*)carve((size_t)KT2 * 32 * 16 * 2);
    float*    h1   = (float*)carve((size_t)NN * HID * 4);
    float*    out1 = (float*)carve((size_t)NN * HID * 4);
    float*    aS1  = (float*)carve((size_t)NN * 8 * 4);
    float*    aD1  = (float*)carve((size_t)NN * 8 * 4);
    unsigned* mk1  = (unsigned*)carve((size_t)NN * 8 * 4);
    float*    dn1  = (float*)carve((size_t)NN * 8 * 4);
    float*    h2   = (float*)carve((size_t)NN * 10 * 4);
    float*    aS2  = (float*)carve((size_t)NN * 4);
    float*    aD2  = (float*)carve((size_t)NN * 4);
    unsigned* mk2  = (unsigned*)carve((size_t)NN * 4);
    float*    dn2  = (float*)carve((size_t)NN * 4);

    // weight swizzle + accumulator init (independent)
    gat_swizzW1<<<(KT1 * 16 * 32 * 16 + 255) / 256, 256, 0, stream>>>(W1, W1b);
    gat_swizzW2<<<(KT2 * 32 * 16 + 255) / 256, 256, 0, stream>>>(W2, W2b);
    gat_initOut1<<<(NN * HID + 255) / 256, 256, 0, stream>>>(out1, b1);
    gat_initSc1<<<(NN * 8 + 255) / 256, 256, 0, stream>>>(mk1, dn1);
    gat_initOut2<<<(NN * 10 + 255) / 256, 256, 0, stream>>>(out, b2);
    gat_initSc2<<<(NN + 255) / 256, 256, 0, stream>>>(mk2, dn2);

    // layer 1
    gat_gemm1<<<(NN + 31) / 32, 256, 0, stream>>>(x, W1b, h1);
    gat_attn1<<<(NN * 8 + 255) / 256, 256, 0, stream>>>(h1, as1, ad1, aS1, aD1);
    gat_edgemax1<<<(ET + 255) / 256, 256, 0, stream>>>(ei, aS1, aD1, mk1);
    gat_edgesum1<<<(ET + 255) / 256, 256, 0, stream>>>(ei, aS1, aD1, mk1, dn1);
    gat_agg1<<<(ET + 7) / 8, 256, 0, stream>>>(ei, aS1, aD1, mk1, dn1, h1, out1);
    gat_elu<<<(NN * HID + 255) / 256, 256, 0, stream>>>(out1);

    // layer 2
    gat_gemm2<<<(NN / 16 + 7) / 8, 256, 0, stream>>>(out1, W2b, h2);
    gat_attn2<<<(NN + 255) / 256, 256, 0, stream>>>(h2, as2, ad2, aS2, aD2);
    gat_edgemax2<<<(ET + 255) / 256, 256, 0, stream>>>(ei, aS2, aD2, mk2);
    gat_edgesum2<<<(ET + 255) / 256, 256, 0, stream>>>(ei, aS2, aD2, mk2, dn2);
    gat_agg2<<<(ET + 255) / 256, 256, 0, stream>>>(ei, aS2, aD2, mk2, dn2, h2, out);
}
